// ParallelMoELayer_12859132084354
// MI455X (gfx1250) — matrix operand order repaired and checked by reading, not compiled, not run on hardware
//
#include <hip/hip_runtime.h>
#include <hip/hip_bf16.h>
#include <math.h>
#include <stdint.h>

typedef __bf16 bf16;
typedef __attribute__((ext_vector_type(16))) __bf16 v16bf;
typedef __attribute__((ext_vector_type(8)))  __bf16 v8bf;
typedef __attribute__((ext_vector_type(8)))  float  v8f;
typedef __attribute__((ext_vector_type(4)))  float  v4f;
typedef __attribute__((ext_vector_type(4)))  int    v4i;

#define AS1 __attribute__((address_space(1)))
#define AS3 __attribute__((address_space(3)))

#if defined(__has_builtin)
#if __has_builtin(__builtin_amdgcn_global_load_async_to_lds_b128)
#define HAVE_ASYNC_LDS 1
#endif
#if __has_builtin(__builtin_amdgcn_s_wait_asynccnt)
#define HAVE_WAIT_ASYNC 1
#endif
#endif
#ifndef HAVE_ASYNC_LDS
#define HAVE_ASYNC_LDS 0
#endif
#ifndef HAVE_WAIT_ASYNC
#define HAVE_WAIT_ASYNC 0
#endif

static constexpr int Bc   = 2;
static constexpr int Sc   = 1024;
static constexpr int Dc   = 1024;
static constexpr int Ec   = 8;
static constexpr int Hc   = 3584;
static constexpr int Nc   = Bc * Sc;      // 2048 tokens
static constexpr int NCAP = Nc + 16 * Ec; // 2176 rows incl. per-expert padding
static constexpr int CHUNK = 512;         // K-chunk staged in LDS
static constexpr int XSTR  = 520;         // LDS row stride (bf16): 1040B = 65*16 -> 16B aligned, bank-spread

#define WMMA_BF16(a, b, c) \
  __builtin_amdgcn_wmma_f32_16x16x32_bf16(false, (a), false, (b), (short)0, (c), false, false)

// ---- A fragment from LDS tile (16-bit A 16x32, contiguous k = 16*half + s) -------------
static __device__ __forceinline__ v16bf make_a(const bf16* Xs, int m, int kb, int half) {
  v8bf a0 = *(const v8bf*)(Xs + m * XSTR + kb + half * 16);
  v8bf a1 = *(const v8bf*)(Xs + m * XSTR + kb + half * 16 + 8);
  return __builtin_shufflevector(a0, a1, 0, 1, 2, 3, 4, 5, 6, 7, 8, 9, 10, 11, 12, 13, 14, 15);
}

// ---- B fragment from a global fp32 weight row (16-bit B 32x16 layout) ---------
static __device__ __forceinline__ v16bf load_b(const float* __restrict__ wrow, int kstart) {
  v16bf b;
#pragma unroll
  for (int j = 0; j < 16; j += 4) {
    v4f f = *(const v4f*)(wrow + kstart + j);
    b[j + 0] = (bf16)f[0];
    b[j + 1] = (bf16)f[1];
    b[j + 2] = (bf16)f[2];
    b[j + 3] = (bf16)f[3];
  }
  return b;
}

// ---- stage a 16 x CHUNK bf16 tile (row stride `srcstr`) into LDS --------------
static __device__ __forceinline__ void stage_tile(bf16* Xs, const bf16* __restrict__ src,
                                                  size_t srcstr, int row0, int ch, int tid) {
#pragma unroll
  for (int i = 0; i < (16 * CHUNK) / (128 * 8); ++i) {
    int le = (i * 128 + tid) * 8;
    int r = le / CHUNK, c = le % CHUNK;
#if HAVE_ASYNC_LDS
    __builtin_amdgcn_global_load_async_to_lds_b128(
        (AS1 v4i*)(src + (size_t)(row0 + r) * srcstr + ch + c),
        (AS3 v4i*)(Xs + r * XSTR + c), 0, 0);
#else
    *(v8bf*)(Xs + r * XSTR + c) = *(const v8bf*)(src + (size_t)(row0 + r) * srcstr + ch + c);
#endif
  }
#if HAVE_ASYNC_LDS
#if HAVE_WAIT_ASYNC
  __builtin_amdgcn_s_wait_asynccnt(0);               // drain this wave's async copies
#else
  asm volatile("s_wait_asynccnt 0" ::: "memory");
#endif
#endif
}

// ============================== init ==========================================
__global__ void moe_init(uint32_t* __restrict__ Xg32, int* __restrict__ token_ids,
                         float* __restrict__ loadbuf, int* __restrict__ counts,
                         int* __restrict__ cursor, float* __restrict__ zsum) {
  int idx = blockIdx.x * blockDim.x + threadIdx.x;
  if (idx < NCAP * Dc / 2) Xg32[idx] = 0u;            // zero Xg (bf16 pairs)
  if (idx < NCAP) token_ids[idx] = -1;
  if (idx < Sc * Ec) loadbuf[idx] = 0.0f;
  if (idx < Ec) { counts[idx] = 0; cursor[idx] = 0; }
  if (idx == 0) zsum[0] = 0.0f;
}

// ============================== router ========================================
__global__ __launch_bounds__(256) void moe_router(
    const float* __restrict__ x, const float* __restrict__ Wr,
    int* __restrict__ counts, int* __restrict__ eidx,
    float* __restrict__ zsum, float* __restrict__ loadbuf) {
  const int n = blockIdx.x;
  const int tid = threadIdx.x;
  __shared__ float red[Ec * 256];
  const float* xp = x + (size_t)n * Dc;
  float p[Ec];
#pragma unroll
  for (int e = 0; e < Ec; ++e) p[e] = 0.0f;
  const int base = tid * 4;
#pragma unroll
  for (int j = 0; j < 4; ++j) {
    float xv = xp[base + j];
#pragma unroll
    for (int e = 0; e < Ec; ++e) p[e] += xv * Wr[e * Dc + base + j];
  }
#pragma unroll
  for (int e = 0; e < Ec; ++e) red[e * 256 + tid] = p[e];
  __syncthreads();
  for (int s = 128; s > 0; s >>= 1) {
    if (tid < s) {
#pragma unroll
      for (int e = 0; e < Ec; ++e) red[e * 256 + tid] += red[e * 256 + tid + s];
    }
    __syncthreads();
  }
  if (tid == 0) {
    float l[Ec];
    float mx = -1e30f;
    int amax = 0;
#pragma unroll
    for (int e = 0; e < Ec; ++e) {
      l[e] = red[e * 256];
      if (l[e] > mx) { mx = l[e]; amax = e; }
    }
    float se = 0.0f;
#pragma unroll
    for (int e = 0; e < Ec; ++e) { l[e] = __expf(l[e] - mx); se += l[e]; }
    const float inv = 1.0f / se;
    const int srow = n % Sc;
#pragma unroll
    for (int e = 0; e < Ec; ++e) atomicAdd(&loadbuf[srow * Ec + e], l[e] * inv);
    atomicAdd(zsum, mx + logf(se));                   // logsumexp
    eidx[n] = amax;
    atomicAdd(&counts[amax], 1);
  }
}

// ============================== finalize (offsets + aux loss) =================
__global__ __launch_bounds__(256) void moe_finalize(
    const int* __restrict__ counts, int* __restrict__ offsets,
    const float* __restrict__ loadbuf, const float* __restrict__ zsum,
    float* __restrict__ out_aux) {
  __shared__ float s1[256], s2[256];
  const int tid = threadIdx.x;
  float sum = 0.0f, sq = 0.0f;
  for (int i = tid; i < Sc * Ec; i += 256) {
    float v = loadbuf[i];
    sum += v;
    sq += v * v;
  }
  s1[tid] = sum; s2[tid] = sq;
  __syncthreads();
  for (int s = 128; s > 0; s >>= 1) {
    if (tid < s) { s1[tid] += s1[tid + s]; s2[tid] += s2[tid + s]; }
    __syncthreads();
  }
  if (tid == 0) {
    int off = 0;
    for (int e = 0; e < Ec; ++e) {
      offsets[e] = off;
      off += (counts[e] + 15) & ~15;                  // 16-aligned groups
    }
    const float Nl = (float)(Sc * Ec);
    float mean = s1[0] / Nl;
    float var = (s2[0] - Nl * mean * mean) / (Nl - 1.0f);  // ddof=1
    float load_loss = var / (mean * mean);            // (std/mean)^2
    float zm = zsum[0] / (float)Nc;
    out_aux[0] = 1e-3f * (zm * zm) + 1e-3f * load_loss;
  }
}

// ============================== gather (fp32 -> bf16, grouped) ================
__global__ __launch_bounds__(256) void moe_gather(
    const float* __restrict__ x, const int* __restrict__ eidx,
    const int* __restrict__ offsets, int* __restrict__ cursor,
    int* __restrict__ token_ids, bf16* __restrict__ Xg) {
  const int n = blockIdx.x;
  __shared__ int slot_s;
  if (threadIdx.x == 0) {
    int e = eidx[n];
    int pos = atomicAdd(&cursor[e], 1);
    int slot = offsets[e] + pos;
    token_ids[slot] = n;
    slot_s = slot;
  }
  __syncthreads();
  const int slot = slot_s;
  const float* xp = x + (size_t)n * Dc;
  bf16* dst = Xg + (size_t)slot * Dc;
  for (int i = threadIdx.x; i < Dc; i += 256) dst[i] = (bf16)xp[i];
}

// ============================== FFN stage 1: silu(xW1^T)*(xW2^T) ==============
// grid = (H/64, N/16, E), block = 128 (4 waves). Wave w covers h-cols [hb+16w, +16).
__global__ __launch_bounds__(128) void moe_ffn1(
    const float* __restrict__ W1, const float* __restrict__ W2,
    const bf16* __restrict__ Xg, bf16* __restrict__ Hbuf,
    const int* __restrict__ counts, const int* __restrict__ offsets) {
  const int e = blockIdx.z;
  if ((int)blockIdx.y * 16 >= counts[e]) return;      // empty tile for this expert
  const int row0 = offsets[e] + blockIdx.y * 16;
  const int hb = blockIdx.x * 64;
  __shared__ bf16 Xs[16 * XSTR];
  const int tid = threadIdx.x;
  const int wave = tid >> 5, lane = tid & 31, m = lane & 15, half = lane >> 4;
  const int h = hb + wave * 16 + m;
  const float* w1row = W1 + ((size_t)e * Hc + h) * Dc;
  const float* w2row = W2 + ((size_t)e * Hc + h) * Dc;
  v8f acc1 = {0.f, 0.f, 0.f, 0.f, 0.f, 0.f, 0.f, 0.f};
  v8f acc2 = acc1;
  for (int ch = 0; ch < Dc; ch += CHUNK) {
    if (ch + CHUNK < Dc) {                            // stream next weight chunk
      __builtin_prefetch(w1row + ch + CHUNK, 0, 1);
      __builtin_prefetch(w2row + ch + CHUNK, 0, 1);
    }
    stage_tile(Xs, Xg, (size_t)Dc, row0, ch, tid);
    __syncthreads();
#pragma unroll 4
    for (int kk = 0; kk < CHUNK; kk += 32) {
      v16bf a = make_a(Xs, m, kk, half);
      int kg = ch + kk + half * 16;
      v16bf b1 = load_b(w1row, kg);
      v16bf b2 = load_b(w2row, kg);
      acc1 = WMMA_BF16(a, b1, acc1);
      acc2 = WMMA_BF16(a, b2, acc2);
    }
    __syncthreads();
  }
  // C layout: elem r -> row M = r + 8*half, col N = m
#pragma unroll
  for (int r = 0; r < 8; ++r) {
    float x1 = acc1[r], x2 = acc2[r];
    float g = (x1 / (1.0f + __expf(-x1))) * x2;       // silu(x1) * x2
    int rowi = row0 + r + half * 8;
    Hbuf[(size_t)rowi * Hc + h] = (bf16)g;
  }
}

// ============================== FFN stage 2: out = G W3^T (scatter) ===========
// grid = (D/64, N/16, E), block = 128 (4 waves).
__global__ __launch_bounds__(128) void moe_ffn2(
    const float* __restrict__ W3, const bf16* __restrict__ Hbuf,
    const int* __restrict__ counts, const int* __restrict__ offsets,
    const int* __restrict__ token_ids, float* __restrict__ out) {
  const int e = blockIdx.z;
  if ((int)blockIdx.y * 16 >= counts[e]) return;
  const int row0 = offsets[e] + blockIdx.y * 16;
  const int db = blockIdx.x * 64;
  __shared__ bf16 Gs[16 * XSTR];
  const int tid = threadIdx.x;
  const int wave = tid >> 5, lane = tid & 31, m = lane & 15, half = lane >> 4;
  const int d = db + wave * 16 + m;
  const float* w3row = W3 + ((size_t)e * Dc + d) * Hc;
  v8f acc = {0.f, 0.f, 0.f, 0.f, 0.f, 0.f, 0.f, 0.f};
  for (int ch = 0; ch < Hc; ch += CHUNK) {
    if (ch + CHUNK < Hc) __builtin_prefetch(w3row + ch + CHUNK, 0, 1);
    stage_tile(Gs, Hbuf, (size_t)Hc, row0, ch, tid);
    __syncthreads();
#pragma unroll 4
    for (int kk = 0; kk < CHUNK; kk += 32) {
      v16bf a = make_a(Gs, m, kk, half);
      v16bf b = load_b(w3row, ch + kk + half * 16);
      acc = WMMA_BF16(a, b, acc);
    }
    __syncthreads();
  }
#pragma unroll
  for (int r = 0; r < 8; ++r) {
    int rowi = row0 + r + half * 8;
    int id = token_ids[rowi];
    if (id >= 0) out[(size_t)id * Dc + d] = acc[r];   // each token written exactly once
  }
}

// ============================== host launcher =================================
extern "C" void kernel_launch(void* const* d_in, const int* in_sizes, int n_in,
                              void* d_out, int out_size, void* d_ws, size_t ws_size,
                              hipStream_t stream) {
  (void)in_sizes; (void)n_in; (void)out_size; (void)ws_size;
  const float* x  = (const float*)d_in[0];
  const float* Wr = (const float*)d_in[1];
  const float* W1 = (const float*)d_in[2];
  const float* W2 = (const float*)d_in[3];
  const float* W3 = (const float*)d_in[4];
  float* out = (float*)d_out;

  uint8_t* ws = (uint8_t*)d_ws;
  size_t off = 0;
  auto alloc = [&](size_t bytes) -> void* {
    void* p = ws + off;
    off = (off + bytes + 255) & ~(size_t)255;
    return p;
  };
  int*   counts    = (int*)alloc(Ec * sizeof(int));
  int*   cursor    = (int*)alloc(Ec * sizeof(int));
  int*   offsets   = (int*)alloc(Ec * sizeof(int));
  int*   eidx      = (int*)alloc(Nc * sizeof(int));
  float* zsum      = (float*)alloc(sizeof(float));
  float* loadbuf   = (float*)alloc((size_t)Sc * Ec * sizeof(float));
  int*   token_ids = (int*)alloc(NCAP * sizeof(int));
  bf16*  Xg        = (bf16*)alloc((size_t)NCAP * Dc * sizeof(bf16));
  bf16*  Hbuf      = (bf16*)alloc((size_t)NCAP * Hc * sizeof(bf16));

  const int init_threads = NCAP * Dc / 2;             // dominates all init ranges
  moe_init<<<(init_threads + 255) / 256, 256, 0, stream>>>(
      (uint32_t*)Xg, token_ids, loadbuf, counts, cursor, zsum);
  moe_router<<<Nc, 256, 0, stream>>>(x, Wr, counts, eidx, zsum, loadbuf);
  moe_finalize<<<1, 256, 0, stream>>>(counts, offsets, loadbuf, zsum,
                                      out + (size_t)Nc * Dc);
  moe_gather<<<Nc, 256, 0, stream>>>(x, eidx, offsets, cursor, token_ids, Xg);
  moe_ffn1<<<dim3(Hc / 64, Nc / 16, Ec), 128, 0, stream>>>(W1, W2, Xg, Hbuf, counts, offsets);
  moe_ffn2<<<dim3(Dc / 64, Nc / 16, Ec), 128, 0, stream>>>(W3, Hbuf, counts, offsets, token_ids, out);
}
